// BasicMoE_83399674953935
// MI455X (gfx1250) — matrix operand order repaired and checked by reading, not compiled, not run on hardware
//
#include <hip/hip_runtime.h>
#include <hip/hip_bf16.h>

// ---------------------------------------------------------------------------
// BasicMoE fused kernels for gfx1250 (MI455X), wave32 + WMMA bf16.
// out[n,o] = sum_e softmax(x@Wg+bg)[n,e] * ( (x @ We[e])[n,o] + be[e,o] )
//
// Precision: split-bf16 (hi/lo) 3-term GEMM (xh@Wh + xh@Wl + xl@Wh) with fp32
// accumulate -> ~fp32 accuracy on the XDL path.
//
// Pipeline:
//   1) convert_x / convert_w: one-pass fp32 -> bf16 hi/lo split, written to
//      d_ws in the exact wave32 WMMA fragment layout (32 contiguous bytes per
//      lane per 16x32 / 32x16 fragment).
//   2) gate kernel: softmax(x@Wg+bg), one wave per token.
//   3) main kernel: LDS-free, barrier-free GEMM. Fragments loaded straight
//      from L2-resident tiled buffers with global_load_b128 (SGPR base +
//      32-bit lane offset, strength-reduced by +2048B/iter), double-buffered
//      against the WMMAs. Gate-weighted fold + bias in fp32 epilogue.
// ---------------------------------------------------------------------------

#define IN_DIM      1024
#define OUT_DIM     1024
#define NUM_EXPERTS 8
#define N_TOKENS    8192

typedef __attribute__((ext_vector_type(16))) __bf16 v16bf;
typedef __attribute__((ext_vector_type(8)))  float  v8f;

union FragBF {
    v16bf v;
    uint4 q[2];
};
union Pack16 {
    unsigned short s[16];
    uint4 q[2];
};

__device__ __forceinline__ unsigned f2bf(float f) {
    unsigned u = __float_as_uint(f);
    return (u + 0x7fffu + ((u >> 16) & 1u)) >> 16;   // RNE
}
__device__ __forceinline__ float bf2f(unsigned h) {
    return __uint_as_float(h << 16);
}

// Workspace layout (bytes)
#define WS_GATES 0
#define WS_XH    (262144)                       // 8192*8*4 gates first
#define WS_XL    (WS_XH + 16777216)             // each split: 8M bf16 = 16MB
#define WS_WH    (WS_XL + 16777216)
#define WS_WL    (WS_WH + 16777216)

// ---------------------------------------------------------------------------
// Kernel 1a: x -> xh/xl, tiled per (mtile, ktile32, lane): 16 bf16 per lane.
//   elem e -> k = kt*32 + 16*half + e                    (A 16x32 bf16 layout)
// ---------------------------------------------------------------------------
__global__ __launch_bounds__(256)
void convert_x_kernel(const float* __restrict__ x,
                      unsigned short* __restrict__ xh,
                      unsigned short* __restrict__ xl) {
    const int wid  = blockIdx.x * 8 + (threadIdx.x >> 5);  // 0..16383
    const int lane = threadIdx.x & 31;
    const int mt   = wid >> 5;     // 0..511
    const int kt   = wid & 31;     // 0..31
    const int half = lane >> 4, l16 = lane & 15;

    const int m  = mt * 16 + l16;
    const int k0 = kt * 32 + 16 * half;
    const int k1 = k0 + 8;
    const float* __restrict__ xr = x + (size_t)m * IN_DIM;

    float4 a0 = *(const float4*)(xr + k0);
    float4 a1 = *(const float4*)(xr + k0 + 4);
    float4 b0 = *(const float4*)(xr + k1);
    float4 b1 = *(const float4*)(xr + k1 + 4);
    float v[16] = { a0.x,a0.y,a0.z,a0.w, a1.x,a1.y,a1.z,a1.w,
                    b0.x,b0.y,b0.z,b0.w, b1.x,b1.y,b1.z,b1.w };

    Pack16 h, l;
#pragma unroll
    for (int e = 0; e < 16; ++e) {
        unsigned hh = f2bf(v[e]);
        h.s[e] = (unsigned short)hh;
        l.s[e] = (unsigned short)f2bf(v[e] - bf2f(hh));
    }
    size_t off = ((size_t)wid * 32 + lane) * 16;   // element index
    *(uint4*)(xh + off)     = h.q[0];
    *(uint4*)(xh + off + 8) = h.q[1];
    *(uint4*)(xl + off)     = l.q[0];
    *(uint4*)(xl + off + 8) = l.q[1];
}

// ---------------------------------------------------------------------------
// Kernel 1b: We -> wh/wl, tiled per (expert, ntile, ktile32, lane):
//   elem e -> k = kt*32 + 16*half + e, n = nt*16 + l16   (B 32x16 bf16 layout)
// ---------------------------------------------------------------------------
__global__ __launch_bounds__(256)
void convert_w_kernel(const float* __restrict__ We,
                      unsigned short* __restrict__ wh,
                      unsigned short* __restrict__ wl) {
    const int wid  = blockIdx.x * 8 + (threadIdx.x >> 5);  // 0..16383
    const int lane = threadIdx.x & 31;
    const int ex   = wid >> 11;          // 0..7
    const int nt   = (wid >> 5) & 63;    // 0..63
    const int kt   = wid & 31;           // 0..31
    const int half = lane >> 4, l16 = lane & 15;

    const int n     = nt * 16 + l16;
    const int kbase = kt * 32 + 16 * half;
    const float* __restrict__ wb = We + (size_t)ex * IN_DIM * OUT_DIM + n;

    Pack16 h, l;
#pragma unroll
    for (int e = 0; e < 16; ++e) {
        float v = wb[(size_t)(kbase + e) * OUT_DIM];
        unsigned hh = f2bf(v);
        h.s[e] = (unsigned short)hh;
        l.s[e] = (unsigned short)f2bf(v - bf2f(hh));
    }
    size_t off = ((size_t)wid * 32 + lane) * 16;
    *(uint4*)(wh + off)     = h.q[0];
    *(uint4*)(wh + off + 8) = h.q[1];
    *(uint4*)(wl + off)     = l.q[0];
    *(uint4*)(wl + off + 8) = l.q[1];
}

// ---------------------------------------------------------------------------
// Kernel 2: gates = softmax(x @ Wg + bg).  One wave32 per token.
// ---------------------------------------------------------------------------
__global__ __launch_bounds__(256)
void moe_gate_kernel(const float* __restrict__ x,
                     const float* __restrict__ Wg,
                     const float* __restrict__ bg,
                     float* __restrict__ gates) {
    const int lane  = threadIdx.x & 31;
    const int wave  = threadIdx.x >> 5;
    const int token = blockIdx.x * 8 + wave;

    const float* __restrict__ xr = x + (size_t)token * IN_DIM;

    float p[NUM_EXPERTS];
#pragma unroll
    for (int e = 0; e < NUM_EXPERTS; ++e) p[e] = 0.f;

    for (int i = lane; i < IN_DIM; i += 32) {
        float xv = xr[i];
        const float4* wr = (const float4*)(Wg + (size_t)i * NUM_EXPERTS);
        float4 w0 = wr[0], w1 = wr[1];
        p[0] += xv * w0.x; p[1] += xv * w0.y; p[2] += xv * w0.z; p[3] += xv * w0.w;
        p[4] += xv * w1.x; p[5] += xv * w1.y; p[6] += xv * w1.z; p[7] += xv * w1.w;
    }
#pragma unroll
    for (int off = 16; off > 0; off >>= 1) {
#pragma unroll
        for (int e = 0; e < NUM_EXPERTS; ++e)
            p[e] += __shfl_xor(p[e], off, 32);
    }
#pragma unroll
    for (int e = 0; e < NUM_EXPERTS; ++e) p[e] += bg[e];

    float mx = p[0];
#pragma unroll
    for (int e = 1; e < NUM_EXPERTS; ++e) mx = fmaxf(mx, p[e]);
    float s = 0.f;
#pragma unroll
    for (int e = 0; e < NUM_EXPERTS; ++e) { p[e] = __expf(p[e] - mx); s += p[e]; }
    float inv = 1.f / s;
    if (lane < NUM_EXPERTS)
        gates[(size_t)token * NUM_EXPERTS + lane] = p[lane] * inv;
}

// ---------------------------------------------------------------------------
// Kernel 3: LDS-free fused all-expert GEMM + gate-weighted sum + bias.
// Block: 256 threads = 8 waves (4 x 2); wave tile 32x32 = 2x2 WMMA tiles.
// Block tile 128(M) x 64(N).  K loop: 32 substeps of 32, double-buffered,
// addresses strength-reduced to 32-bit offsets + immediate offsets.
// ---------------------------------------------------------------------------
#define BM 128
#define BN 64

__global__ __launch_bounds__(256)
void moe_main_kernel(const unsigned short* __restrict__ xh_,
                     const unsigned short* __restrict__ xl_,
                     const unsigned short* __restrict__ wh_,
                     const unsigned short* __restrict__ wl_,
                     const float* __restrict__ be,
                     const float* __restrict__ gates,
                     float* __restrict__ out) {
    __shared__ float gS[BM * NUM_EXPERTS];

    const char* __restrict__ xh = (const char*)xh_;
    const char* __restrict__ xl = (const char*)xl_;
    const char* __restrict__ wh = (const char*)wh_;
    const char* __restrict__ wl = (const char*)wl_;

    const int t    = threadIdx.x;
    const int lane = t & 31;
    const int wave = t >> 5;
    const int wm   = wave >> 1;        // 0..3
    const int wn   = wave & 1;         // 0..1
    const int half = lane >> 4;
    const int l16  = lane & 15;
    const int bm   = blockIdx.x * BM;
    const int bn   = blockIdx.y * BN;

    // stage this block's gates (128 tokens x 8 experts) once
    for (int i = t; i < BM * NUM_EXPERTS; i += 256)
        gS[i] = gates[(size_t)(bm + (i >> 3)) * NUM_EXPERTS + (i & 7)];
    __syncthreads();

    const int mtA[2] = { (bm >> 4) + wm * 2 + 0, (bm >> 4) + wm * 2 + 1 };
    const int ntB[2] = { (bn >> 4) + wn * 2 + 0, (bn >> 4) + wn * 2 + 1 };
    const unsigned laneB = (unsigned)(lane << 5);   // lane * 32 bytes

    const v8f vzero = {};
    v8f facc[2][2];
#pragma unroll
    for (int a = 0; a < 2; ++a)
#pragma unroll
        for (int b = 0; b < 2; ++b) facc[a][b] = vzero;

    for (int ex = 0; ex < NUM_EXPERTS; ++ex) {
        v8f acc[2][2];
#pragma unroll
        for (int a = 0; a < 2; ++a)
#pragma unroll
            for (int b = 0; b < 2; ++b) acc[a][b] = vzero;

        // 32-bit running byte offsets; one per A row-tile / B col-tile.
        // Hi and Lo buffers share the same offsets (different SGPR bases).
        unsigned aoff0 = ((unsigned)(mtA[0] * 32) << 10) + laneB;
        unsigned aoff1 = ((unsigned)(mtA[1] * 32) << 10) + laneB;
        unsigned boff0 = ((unsigned)((ex * 64 + ntB[0]) * 32) << 10) + laneB;
        unsigned boff1 = ((unsigned)((ex * 64 + ntB[1]) * 32) << 10) + laneB;

        // frag sets: [buf][ ah0, ah1, al0, al1, bh0, bh1, bl0, bl1 ]
        FragBF fr[2][8];

        // IMM = compile-time byte displacement (folds into 24-bit ioffset).
        auto load_set = [&](int buf, int IMM) {
            fr[buf][0].q[0] = *(const uint4*)(xh + aoff0 + IMM);
            fr[buf][0].q[1] = *(const uint4*)(xh + aoff0 + IMM + 16);
            fr[buf][1].q[0] = *(const uint4*)(xh + aoff1 + IMM);
            fr[buf][1].q[1] = *(const uint4*)(xh + aoff1 + IMM + 16);
            fr[buf][2].q[0] = *(const uint4*)(xl + aoff0 + IMM);
            fr[buf][2].q[1] = *(const uint4*)(xl + aoff0 + IMM + 16);
            fr[buf][3].q[0] = *(const uint4*)(xl + aoff1 + IMM);
            fr[buf][3].q[1] = *(const uint4*)(xl + aoff1 + IMM + 16);
            fr[buf][4].q[0] = *(const uint4*)(wh + boff0 + IMM);
            fr[buf][4].q[1] = *(const uint4*)(wh + boff0 + IMM + 16);
            fr[buf][5].q[0] = *(const uint4*)(wh + boff1 + IMM);
            fr[buf][5].q[1] = *(const uint4*)(wh + boff1 + IMM + 16);
            fr[buf][6].q[0] = *(const uint4*)(wl + boff0 + IMM);
            fr[buf][6].q[1] = *(const uint4*)(wl + boff0 + IMM + 16);
            fr[buf][7].q[0] = *(const uint4*)(wl + boff1 + IMM);
            fr[buf][7].q[1] = *(const uint4*)(wl + boff1 + IMM + 16);
        };
        auto mm = [&](int buf) {
#pragma unroll
            for (int rt = 0; rt < 2; ++rt)
#pragma unroll
                for (int ct = 0; ct < 2; ++ct) {
                    acc[rt][ct] = __builtin_amdgcn_wmma_f32_16x16x32_bf16(
                        false, fr[buf][rt].v,     false, fr[buf][4 + ct].v,
                        (short)0, acc[rt][ct], false, false);
                    acc[rt][ct] = __builtin_amdgcn_wmma_f32_16x16x32_bf16(
                        false, fr[buf][rt].v,     false, fr[buf][6 + ct].v,
                        (short)0, acc[rt][ct], false, false);
                    acc[rt][ct] = __builtin_amdgcn_wmma_f32_16x16x32_bf16(
                        false, fr[buf][2 + rt].v, false, fr[buf][4 + ct].v,
                        (short)0, acc[rt][ct], false, false);
                }
        };

        load_set(0, 0);                       // kt = 0
        for (int kt = 0; kt < 30; kt += 2) {  // steady state: no conditionals
            load_set(1, 1024);                // kt + 1
            mm(0);
            load_set(0, 2048);                // kt + 2
            mm(1);
            aoff0 += 2048; aoff1 += 2048;
            boff0 += 2048; boff1 += 2048;
        }
        load_set(1, 1024);                    // kt = 31 (tail)
        mm(0);
        mm(1);

        // fold expert ex: facc += g * (acc + be)
#pragma unroll
        for (int rt = 0; rt < 2; ++rt) {
            int m0 = wm * 32 + rt * 16 + 8 * half;   // C/D layout: row = r + 8*half
#pragma unroll
            for (int ct = 0; ct < 2; ++ct) {
                int gn    = bn + wn * 32 + ct * 16 + l16;
                float bev = be[(size_t)ex * OUT_DIM + gn];
#pragma unroll
                for (int r = 0; r < 8; ++r) {
                    float g = gS[(m0 + r) * NUM_EXPERTS + ex];
                    facc[rt][ct][r] += g * (acc[rt][ct][r] + bev);
                }
            }
        }
    }

    // epilogue: fp32 output
#pragma unroll
    for (int rt = 0; rt < 2; ++rt) {
        int gm0 = bm + wm * 32 + rt * 16 + 8 * half;
#pragma unroll
        for (int ct = 0; ct < 2; ++ct) {
            int gn = bn + wn * 32 + ct * 16 + l16;
#pragma unroll
            for (int r = 0; r < 8; ++r)
                out[(size_t)(gm0 + r) * OUT_DIM + gn] = facc[rt][ct][r];
        }
    }
}

// ---------------------------------------------------------------------------
extern "C" void kernel_launch(void* const* d_in, const int* in_sizes, int n_in,
                              void* d_out, int out_size, void* d_ws, size_t ws_size,
                              hipStream_t stream) {
    const float* x  = (const float*)d_in[0];
    const float* We = (const float*)d_in[1];
    const float* be = (const float*)d_in[2];
    const float* Wg = (const float*)d_in[3];
    const float* bg = (const float*)d_in[4];
    float* out = (float*)d_out;

    char* ws = (char*)d_ws;
    float*          gates = (float*)(ws + WS_GATES);
    unsigned short* xh    = (unsigned short*)(ws + WS_XH);
    unsigned short* xl    = (unsigned short*)(ws + WS_XL);
    unsigned short* wh    = (unsigned short*)(ws + WS_WH);
    unsigned short* wl    = (unsigned short*)(ws + WS_WL);

    convert_x_kernel<<<2048, 256, 0, stream>>>(x, xh, xl);
    convert_w_kernel<<<2048, 256, 0, stream>>>(We, wh, wl);
    moe_gate_kernel<<<N_TOKENS / 8, 256, 0, stream>>>(x, Wg, bg, gates);
    moe_main_kernel<<<dim3(N_TOKENS / BM, OUT_DIM / BN), 256, 0, stream>>>(
        xh, xl, wh, wl, be, gates, out);
}
